// CausalSelfAttention_59416577573383
// MI455X (gfx1250) — matrix operand order repaired and checked by reading, not compiled, not run on hardware
//
#include <hip/hip_runtime.h>

typedef _Float16 v8h  __attribute__((ext_vector_type(8)));
typedef _Float16 v16h __attribute__((ext_vector_type(16)));
typedef float    v8f  __attribute__((ext_vector_type(8)));

#define WMMA(a, b, c) \
  __builtin_amdgcn_wmma_f32_16x16x32_f16(false, (a), false, (b), (short)0, (c), false, false)

static constexpr int BB = 4;      // batch
static constexpr int T  = 2048;   // seq len
static constexpr int C  = 1024;   // model dim
static constexpr int H  = 16;     // heads
static constexpr int Dh = 64;     // head dim
static constexpr int M_TOK = BB * T;  // 8192 tokens

// ---------------------------------------------------------------------------
// Fragment loaders (wave32, contiguous k order k = 16*(lane/16) + s, same for A and B)
// A fragment: 16 rows x 32 k, source row-major [rows, ld]
//   lanes 0-15: row = lane, k = 0..15; lanes 16-31: k = 16..31
// ---------------------------------------------------------------------------
__device__ __forceinline__ v16h load_fragA(const _Float16* base, int row, int ld,
                                           int col, int lane) {
  const _Float16* p =
      base + (size_t)(row + (lane & 15)) * ld + col + ((lane >> 4) << 4);
  v8h lo = *(const v8h*)p;
  v8h hi = *(const v8h*)(p + 8);
  v16h r;
#pragma unroll
  for (int i = 0; i < 8; ++i) { r[i] = lo[i]; r[i + 8] = hi[i]; }
  return r;
}

// B fragment: 32 k x 16 cols; source is B^T row-major [cols, K] so each lane
// reads 16 contiguous halves along k.
//   lanes 0-15: col = lane, k = 0..15 ; lanes 16-31: col = lane-16, k = 16..31
__device__ __forceinline__ v16h load_fragB(const _Float16* base, int nrow, int ld,
                                           int col, int lane) {
  const _Float16* p =
      base + (size_t)(nrow + (lane & 15)) * ld + col + ((lane >> 4) << 4);
  return *(const v16h*)p;
}

// ---------------------------------------------------------------------------
// Elementwise casts
// ---------------------------------------------------------------------------
__global__ void k_cast_f16(const float* __restrict__ in, _Float16* __restrict__ out,
                           int n) {
  int i = blockIdx.x * 256 + threadIdx.x;
  if (i < n) out[i] = (_Float16)in[i];
}

// in: [K, N] row-major f32 ; out: [N, K] row-major f16  (out[n*K+k] = in[k*N+n])
__global__ void k_transpose_cast(const float* __restrict__ in,
                                 _Float16* __restrict__ out, int K, int N) {
  int i = blockIdx.x * 256 + threadIdx.x;
  if (i < K * N) {
    int n = i / K;
    int k = i - n * K;
    out[i] = (_Float16)in[(size_t)k * N + n];
  }
}

// ---------------------------------------------------------------------------
// QKV projection GEMM:  qkv[8192,3072] = xh[8192,1024] @ wqkvT^T + bias
// 8 waves / WG, wave tile 32x64 (2x4 wmma tiles), WG tile 128x128.
// Scatter-stores Q,K in [B,H,T,D] f16 and V transposed in [B,H,D,T] f16.
// ---------------------------------------------------------------------------
__global__ void __launch_bounds__(256)
k_gemm_qkv(const _Float16* __restrict__ A, const _Float16* __restrict__ Bt,
           const float* __restrict__ bias, _Float16* __restrict__ qh,
           _Float16* __restrict__ kh, _Float16* __restrict__ vt) {
  int lane  = threadIdx.x & 31;
  int wave  = threadIdx.x >> 5;
  int mBase = blockIdx.x * 128 + (wave >> 1) * 32;
  int nBase = blockIdx.y * 128 + (wave & 1) * 64;

  v8f acc[2][4] = {};
  for (int k = 0; k < C; k += 32) {
    v16h a0 = load_fragA(A, mBase,      C, k, lane);
    v16h a1 = load_fragA(A, mBase + 16, C, k, lane);
    v16h b0 = load_fragB(Bt, nBase,      C, k, lane);
    v16h b1 = load_fragB(Bt, nBase + 16, C, k, lane);
    v16h b2 = load_fragB(Bt, nBase + 32, C, k, lane);
    v16h b3 = load_fragB(Bt, nBase + 48, C, k, lane);
    acc[0][0] = WMMA(a0, b0, acc[0][0]);
    acc[0][1] = WMMA(a0, b1, acc[0][1]);
    acc[0][2] = WMMA(a0, b2, acc[0][2]);
    acc[0][3] = WMMA(a0, b3, acc[0][3]);
    acc[1][0] = WMMA(a1, b0, acc[1][0]);
    acc[1][1] = WMMA(a1, b1, acc[1][1]);
    acc[1][2] = WMMA(a1, b2, acc[1][2]);
    acc[1][3] = WMMA(a1, b3, acc[1][3]);
  }

  int ln = lane & 15, hb = lane >> 4;
#pragma unroll
  for (int i = 0; i < 2; ++i) {
#pragma unroll
    for (int j = 0; j < 4; ++j) {
      int tn  = nBase + j * 16 + ln;     // 0..3071
      int sel = tn >> 10;                // 0=q 1=k 2=v
      int cc  = tn & 1023;
      int h   = cc >> 6, d = cc & 63;
      float bb = bias[tn];
#pragma unroll
      for (int e = 0; e < 8; ++e) {
        int tm = mBase + i * 16 + e + (hb << 3);  // 0..8191
        int bi = tm >> 11, t = tm & 2047;
        _Float16 hv = (_Float16)(acc[i][j][e] + bb);
        size_t bhidx = (size_t)(bi * H + h);
        if (sel == 0)      qh[(bhidx * T + t) * Dh + d] = hv;
        else if (sel == 1) kh[(bhidx * T + t) * Dh + d] = hv;
        else               vt[(bhidx * Dh + d) * T + t] = hv;  // V transposed
      }
    }
  }
}

// ---------------------------------------------------------------------------
// Flash attention. grid = (T/128, B*H), 256 threads = 8 waves; each wave owns
// 16 query rows. Computes S^T = K.Q^T per 32-wide k block (softmax reduction
// mostly within-lane), then O^T += V^T.P^T with P^T fed from registers.
// ---------------------------------------------------------------------------
__global__ void __launch_bounds__(256)
k_attn(const _Float16* __restrict__ Q, const _Float16* __restrict__ K,
       const _Float16* __restrict__ Vt, _Float16* __restrict__ Oa) {
  int lane = threadIdx.x & 31;
  int wave = threadIdx.x >> 5;
  int ln = lane & 15, hb = lane >> 4;
  int bh = blockIdx.y;                 // b*H + h
  int b = bh >> 4, h = bh & 15;
  const _Float16* Qp = Q  + (size_t)bh * T * Dh;
  const _Float16* Kp = K  + (size_t)bh * T * Dh;
  const _Float16* Vp = Vt + (size_t)bh * Dh * T;   // [Dh, T]
  int qbase = blockIdx.x * 128 + wave * 16;
  int qglob = qbase + ln;

  // Q as B-operand (column n of Q^T = row qbase+n of Q, contiguous in d)
  v16h qB0 = load_fragB(Qp, qbase, Dh, 0,  lane);
  v16h qB1 = load_fragB(Qp, qbase, Dh, 32, lane);

  v8f o0 = {}, o1 = {}, o2 = {}, o3 = {};
  float m = -1e30f, l = 0.0f;
  const float scale = 0.125f;          // 1/sqrt(64)

  for (int kb = 0; kb < qbase + 16; kb += 32) {
    // S^T tiles: st0 covers k rows kb..kb+15, st1 covers kb+16..kb+31
    v8f st0 = {}, st1 = {};
    st0 = WMMA(load_fragA(Kp, kb,      Dh, 0,  lane), qB0, st0);
    st0 = WMMA(load_fragA(Kp, kb,      Dh, 32, lane), qB1, st0);
    st1 = WMMA(load_fragA(Kp, kb + 16, Dh, 0,  lane), qB0, st1);
    st1 = WMMA(load_fragA(Kp, kb + 16, Dh, 32, lane), qB1, st1);

    // D-layout: lane holds q = ln; element e -> k = kb + 8*hb + e (st0), +16 (st1)
    float s0[8], s1[8];
    float rmax = -1e30f;
#pragma unroll
    for (int e = 0; e < 8; ++e) {
      int k0 = kb + (hb << 3) + e;
      float a = st0[e] * scale; if (k0 > qglob)      a = -1e30f;  // causal mask
      float c = st1[e] * scale; if (k0 + 16 > qglob) c = -1e30f;
      s0[e] = a; s1[e] = c;
      rmax = fmaxf(rmax, fmaxf(a, c));
    }
    rmax = fmaxf(rmax, __shfl_xor(rmax, 16, 32));   // combine lane halves (same q)
    float mnew  = fmaxf(m, rmax);
    float alpha = __expf(m - mnew);
    float p0[8], p1[8], rsum = 0.0f;
#pragma unroll
    for (int e = 0; e < 8; ++e) {
      p0[e] = __expf(s0[e] - mnew);
      p1[e] = __expf(s1[e] - mnew);
      rsum += p0[e] + p1[e];
    }
    rsum += __shfl_xor(rsum, 16, 32);
    l = l * alpha + rsum;
    m = mnew;

    // Build P^T B-fragment (k x q). B wants lane<16: k=kb+0..15, lane>=16:
    // k=kb+16..31; our halves hold k offsets {0..7}/{8..15} -> one xor-16 swap.
    v16h pb;
#pragma unroll
    for (int e = 0; e < 8; ++e) {
      float t0 = __shfl_xor(p0[e], 16, 32);
      float t1 = __shfl_xor(p1[e], 16, 32);
      pb[e]     = (_Float16)(hb ? t1 : p0[e]);
      pb[e + 8] = (_Float16)(hb ? p1[e] : t0);
    }

#pragma unroll
    for (int e = 0; e < 8; ++e) {
      o0[e] *= alpha; o1[e] *= alpha; o2[e] *= alpha; o3[e] *= alpha;
    }
    // O^T[d, q] += V^T[d, k] . P^T[k, q] ; V^T rows contiguous (V stored [Dh,T])
    o0 = WMMA(load_fragA(Vp, 0,  T, kb, lane), pb, o0);
    o1 = WMMA(load_fragA(Vp, 16, T, kb, lane), pb, o1);
    o2 = WMMA(load_fragA(Vp, 32, T, kb, lane), pb, o2);
    o3 = WMMA(load_fragA(Vp, 48, T, kb, lane), pb, o3);
  }

  float inv = 1.0f / l;
  size_t rowoff = ((size_t)b * T + qglob) * C + h * Dh;  // [B,T,C] f16
#pragma unroll
  for (int e = 0; e < 8; ++e) {
    int d = e + (hb << 3);
    Oa[rowoff + d]      = (_Float16)(o0[e] * inv);
    Oa[rowoff + d + 16] = (_Float16)(o1[e] * inv);
    Oa[rowoff + d + 32] = (_Float16)(o2[e] * inv);
    Oa[rowoff + d + 48] = (_Float16)(o3[e] * inv);
  }
}

// ---------------------------------------------------------------------------
// Output projection: out[8192,1024] = attn[8192,1024] @ woutT^T + b_out (f32)
// ---------------------------------------------------------------------------
__global__ void __launch_bounds__(256)
k_gemm_out(const _Float16* __restrict__ A, const _Float16* __restrict__ Bt,
           const float* __restrict__ bias, float* __restrict__ Out) {
  int lane  = threadIdx.x & 31;
  int wave  = threadIdx.x >> 5;
  int mBase = blockIdx.x * 128 + (wave >> 1) * 32;
  int nBase = blockIdx.y * 128 + (wave & 1) * 64;

  v8f acc[2][4] = {};
  for (int k = 0; k < C; k += 32) {
    v16h a0 = load_fragA(A, mBase,      C, k, lane);
    v16h a1 = load_fragA(A, mBase + 16, C, k, lane);
    v16h b0 = load_fragB(Bt, nBase,      C, k, lane);
    v16h b1 = load_fragB(Bt, nBase + 16, C, k, lane);
    v16h b2 = load_fragB(Bt, nBase + 32, C, k, lane);
    v16h b3 = load_fragB(Bt, nBase + 48, C, k, lane);
    acc[0][0] = WMMA(a0, b0, acc[0][0]);
    acc[0][1] = WMMA(a0, b1, acc[0][1]);
    acc[0][2] = WMMA(a0, b2, acc[0][2]);
    acc[0][3] = WMMA(a0, b3, acc[0][3]);
    acc[1][0] = WMMA(a1, b0, acc[1][0]);
    acc[1][1] = WMMA(a1, b1, acc[1][1]);
    acc[1][2] = WMMA(a1, b2, acc[1][2]);
    acc[1][3] = WMMA(a1, b3, acc[1][3]);
  }

  int ln = lane & 15, hb = lane >> 4;
#pragma unroll
  for (int i = 0; i < 2; ++i) {
#pragma unroll
    for (int j = 0; j < 4; ++j) {
      int tn = nBase + j * 16 + ln;
      float bb = bias[tn];
#pragma unroll
      for (int e = 0; e < 8; ++e) {
        int tm = mBase + i * 16 + e + (hb << 3);
        Out[(size_t)tm * C + tn] = acc[i][j][e] + bb;
      }
    }
  }
}

// ---------------------------------------------------------------------------
extern "C" void kernel_launch(void* const* d_in, const int* in_sizes, int n_in,
                              void* d_out, int out_size, void* d_ws, size_t ws_size,
                              hipStream_t stream) {
  const float* x     = (const float*)d_in[0];  // [B,T,C]
  const float* w_qkv = (const float*)d_in[1];  // [C, 3C]
  const float* b_qkv = (const float*)d_in[2];  // [3C]
  const float* w_out = (const float*)d_in[3];  // [C, C]
  const float* b_out = (const float*)d_in[4];  // [C]
  float* out = (float*)d_out;                  // [B,T,C] f32

  // Workspace carve-up (all sizes in f16 elements; total ~87 MB)
  _Float16* ws    = (_Float16*)d_ws;
  _Float16* xh    = ws;                          // 8192*1024
  _Float16* wqkvT = xh    + (size_t)M_TOK * C;   // 3072*1024  (transposed)
  _Float16* woutT = wqkvT + (size_t)3 * C * C;   // 1024*1024  (transposed)
  _Float16* qh    = woutT + (size_t)C * C;       // [B,H,T,D]
  _Float16* kh    = qh    + (size_t)M_TOK * C;   // [B,H,T,D]
  _Float16* vt    = kh    + (size_t)M_TOK * C;   // [B,H,D,T]
  _Float16* attn  = vt    + (size_t)M_TOK * C;   // [B,T,C]

  int nx = M_TOK * C;  // 8388608
  k_cast_f16<<<(nx + 255) / 256, 256, 0, stream>>>(x, xh, nx);
  k_transpose_cast<<<(3 * C * C) / 256, 256, 0, stream>>>(w_qkv, wqkvT, C, 3 * C);
  k_transpose_cast<<<(C * C) / 256, 256, 0, stream>>>(w_out, woutT, C, C);

  k_gemm_qkv<<<dim3(M_TOK / 128, (3 * C) / 128), 256, 0, stream>>>(
      xh, wqkvT, b_qkv, qh, kh, vt);

  k_attn<<<dim3(T / 128, BB * H), 256, 0, stream>>>(qh, kh, vt, attn);

  k_gemm_out<<<dim3(M_TOK / 128, C / 128), 256, 0, stream>>>(
      attn, woutT, b_out, out);
}
